// EventTemplateBank_57681410786056
// MI455X (gfx1250) — matrix operand order repaired and checked by reading, not compiled, not run on hardware
//
#include <hip/hip_runtime.h>

// ---------------------------------------------------------------------------
// out[b,t,e] = (1/480) * sum_{m'=0..79, f=0..5} x[b, t-39+m', f] * D[e,f,79-m']
// D = delay-shifted templates.  K ordered k = m'*8 + f (F padded 6->8, pad=0),
// so an A row is a contiguous window of padded-bf16 x samples (16 B / sample).
// B is prebuilt in per-lane WMMA fragment layout.
//
// K-steps walk in 4 strided phases (kk = p, p+4, ..., p+16); the sliding-
// window identity  fragment(kk+4, sub) == fragment(kk, sub+1)  lets a rolling
// register window of 8 A fragments feed 8 sub-tiles with ONE new A fragment
// and ONE B fragment per step: 4 ds_load_b128 per 8 WMMAs (0.5 ds/WMMA).
// ---------------------------------------------------------------------------

typedef __bf16 v16bf __attribute__((ext_vector_type(16)));
typedef float  v8f   __attribute__((ext_vector_type(8)));

#define S_LEN       32768
#define F_IN        6
#define E_N         16
#define L_T         80
#define KSTEPS      20          // K = 80*8 = 640 -> 20 steps of 32
#define T_PER_BLOCK 1024
#define T_PER_WAVE  128         // 8 sub-tiles of 16 rows per wave
#define NWIN        1103        // samples staged per block: 1024 + 79

union FragU { uint4 q[2]; v16bf v; };

static __device__ inline unsigned pk_bf16(float a, float b) {
  unsigned short lo = __builtin_bit_cast(unsigned short, (__bf16)a);
  unsigned short hi = __builtin_bit_cast(unsigned short, (__bf16)b);
  return (unsigned)lo | ((unsigned)hi << 16);
}

#define WMMA_BF16(A, B, C) \
  __builtin_amdgcn_wmma_f32_16x16x32_bf16(false, (A).v, false, (B).v, (short)0, (C), false, false)

// -------- prep: build B fragments (delay, flip, scale, bf16, frag layout) ----
// 16-bit dense B 32x16 layout (table order): lanes 0-15 hold K=0..7,16..23,
// lanes 16-31 hold K=8..15,24..31 (matches A sample-record order).
// bfrag[(kk*32+lane)*2 + half] = uint4 of 4 packed dwords (8 K values).
__global__ void prep_bfrag(const float* __restrict__ tmpl,
                           const float* __restrict__ delays,
                           uint4* __restrict__ bfrag) {
  int idx = blockIdx.x * blockDim.x + threadIdx.x;
  if (idx >= KSTEPS * 32 * 2) return;
  int kk   = idx >> 6;
  int rem  = idx & 63;
  int lane = rem >> 1;
  int half = rem & 1;
  int e = lane & 15;
  int g = lane >> 4;
  int kbase = kk * 32 + half * 16 + g * 8;
  unsigned dw[4];
#pragma unroll
  for (int j = 0; j < 4; ++j) {
    float v01[2];
#pragma unroll
    for (int p = 0; p < 2; ++p) {
      int K = kbase + 2 * j + p;
      int mprime = K >> 3;
      int f = K & 7;
      float val = 0.0f;
      if (f < F_IN) {
        float dl = delays[e * F_IN + f];
        dl = fminf(fmaxf(dl, -10.0f), 10.0f);
        int d = (int)rintf(dl);                 // round-half-even like jnp.round
        int src = (L_T - 1 - mprime) - d;       // flip folded in
        if (src >= 0 && src < L_T)
          val = tmpl[(e * F_IN + f) * L_T + src] * (1.0f / 480.0f);
      }
      v01[p] = val;
    }
    dw[j] = pk_bf16(v01[0], v01[1]);
  }
  bfrag[idx] = make_uint4(dw[0], dw[1], dw[2], dw[3]);
}

// -------- main: sliding-window GEMM via bf16 WMMA ---------------------------
__global__ void __launch_bounds__(256) corr_wmma(
    const float* __restrict__ x, const uint4* __restrict__ bfrag,
    float* __restrict__ out, int nS) {
  __shared__ uint4 sB[KSTEPS * 32 * 2];   // 20480 B
  __shared__ uint4 sWin[NWIN + 1];        // one 16B record per padded sample

  const int b   = blockIdx.y;
  const int T0  = blockIdx.x * T_PER_BLOCK;
  const int tid = threadIdx.x;

  // B fragments global -> LDS (broadcast data, hot in L2)
  for (int i = tid; i < KSTEPS * 32 * 2; i += 256) sB[i] = bfrag[i];

  // x window -> LDS as bf16, F padded to 8, zero outside [0,S)
  for (int wi = tid; wi < NWIN; wi += 256) {
    int s = T0 - 39 + wi;
    uint4 v = make_uint4(0u, 0u, 0u, 0u);
    if (s >= 0 && s < nS) {
      const float* xs = x + ((size_t)b * nS + s) * F_IN;
      v.x = pk_bf16(xs[0], xs[1]);
      v.y = pk_bf16(xs[2], xs[3]);
      v.z = pk_bf16(xs[4], xs[5]);
    }
    sWin[wi] = v;
  }
  __syncthreads();

  const int lane  = tid & 31;
  const int wave  = tid >> 5;
  const int r     = lane & 15;
  const int g     = lane >> 4;
  const int base0 = wave * T_PER_WAVE + r;   // A-row base (window index units)

  v8f acc0 = {}, acc1 = {}, acc2 = {}, acc3 = {};
  v8f acc4 = {}, acc5 = {}, acc6 = {}, acc7 = {};

  for (int p = 0; p < 4; ++p) {
    // Phase p handles kk = p, p+4, p+8, p+12, p+16.  Fragment G_q (q = 0..11)
    // covers samples base0 + 4*p + 16*q + g (+2).  Step s uses G_s..G_{s+7}
    // for sub-tiles 0..7 -> rolling register window of 8 fragments.
    const int ab = base0 + 4 * p + g;
    FragU A0, A1, A2, A3, A4, A5, A6;
    A0.q[0] = sWin[ab];       A0.q[1] = sWin[ab + 2];
    A1.q[0] = sWin[ab + 16];  A1.q[1] = sWin[ab + 18];
    A2.q[0] = sWin[ab + 32];  A2.q[1] = sWin[ab + 34];
    A3.q[0] = sWin[ab + 48];  A3.q[1] = sWin[ab + 50];
    A4.q[0] = sWin[ab + 64];  A4.q[1] = sWin[ab + 66];
    A5.q[0] = sWin[ab + 80];  A5.q[1] = sWin[ab + 82];
    A6.q[0] = sWin[ab + 96];  A6.q[1] = sWin[ab + 98];
#pragma unroll
    for (int s = 0; s < 5; ++s) {
      const int kk = p + 4 * s;
      FragU bf;
      bf.q[0] = sB[(kk * 32 + lane) * 2 + 0];
      bf.q[1] = sB[(kk * 32 + lane) * 2 + 1];
      FragU A7;
      const int a7 = ab + 16 * (s + 7);
      A7.q[0] = sWin[a7];  A7.q[1] = sWin[a7 + 2];
      acc0 = WMMA_BF16(A0, bf, acc0);
      acc1 = WMMA_BF16(A1, bf, acc1);
      acc2 = WMMA_BF16(A2, bf, acc2);
      acc3 = WMMA_BF16(A3, bf, acc3);
      acc4 = WMMA_BF16(A4, bf, acc4);
      acc5 = WMMA_BF16(A5, bf, acc5);
      acc6 = WMMA_BF16(A6, bf, acc6);
      acc7 = WMMA_BF16(A7, bf, acc7);
      A0 = A1; A1 = A2; A2 = A3; A3 = A4;   // register rotation
      A4 = A5; A5 = A6; A6 = A7;            // (renamed by full unroll)
    }
  }

  // C/D layout: VGPR v holds rows M=v (lanes 0-15) and M=v+8 (lanes 16-31),
  // N = lane&15.  Store coalesced; force out[:, S-1, :] = 0 (reference pad).
  const int tt = T0 + wave * T_PER_WAVE;
  v8f accs[8] = {acc0, acc1, acc2, acc3, acc4, acc5, acc6, acc7};
#pragma unroll
  for (int sub = 0; sub < 8; ++sub) {
#pragma unroll
    for (int v = 0; v < 8; ++v) {
      int t = tt + sub * 16 + v + 8 * g;
      float val = (t == nS - 1) ? 0.0f : accs[sub][v];
      out[((size_t)b * nS + t) * E_N + r] = val;
    }
  }
}

extern "C" void kernel_launch(void* const* d_in, const int* in_sizes, int n_in,
                              void* d_out, int out_size, void* d_ws, size_t ws_size,
                              hipStream_t stream) {
  const float* x      = (const float*)d_in[0];
  const float* tmpl   = (const float*)d_in[1];
  const float* delays = (const float*)d_in[2];
  float* out   = (float*)d_out;
  uint4* bfrag = (uint4*)d_ws;                 // needs 20480 B of workspace

  const int nS = S_LEN;
  const int B  = in_sizes[0] / (nS * F_IN);

  hipLaunchKernelGGL(prep_bfrag, dim3((KSTEPS * 32 * 2 + 255) / 256), dim3(256),
                     0, stream, tmpl, delays, bfrag);

  dim3 grid(nS / T_PER_BLOCK, B);              // (32, 64)
  hipLaunchKernelGGL(corr_wmma, grid, dim3(256), 0, stream, x, bfrag, out, nS);
}
